// ModularCausalDynamics_32083405701253
// MI455X (gfx1250) — matrix operand order repaired and checked by reading, not compiled, not run on hardware
//
#include <hip/hip_runtime.h>
#include <hip/hip_bf16.h>

// ModularCausalDynamics on MI455X (gfx1250, wave32, WMMA bf16).
// B=64, K=64, D=128, H=128, T=4, NUM_MP=2.
// Workspace usage: ~30.1 MB.

#define Bq 64
#define Kq 64
#define Dq 128
#define Hq 128
#define Tq 4
#define ROWS (Bq * Kq)            // 4096
#define LDK 136                   // padded bf16 LDS stride (breaks 256B bank period)

typedef __bf16 bf16_t;
typedef __attribute__((ext_vector_type(16))) __bf16 v16bf;
typedef __attribute__((ext_vector_type(8)))  __bf16 v8bf;
typedef __attribute__((ext_vector_type(4)))  __bf16 v4bf;
typedef __attribute__((ext_vector_type(8)))  float  v8f;

__device__ inline bf16_t f2bf(float f) { return (bf16_t)f; }

__device__ inline void store8_bf16(bf16_t* dst, const float* v) {
    v8bf o;
#pragma unroll
    for (int i = 0; i < 8; ++i) o[i] = (bf16_t)v[i];
    *(v8bf*)dst = o;                       // one ds_store_b128
}
__device__ inline void store4_bf16(bf16_t* dst, const float* v) {
    v4bf o;
#pragma unroll
    for (int i = 0; i < 4; ++i) o[i] = (bf16_t)v[i];
    *(v4bf*)dst = o;                       // one ds_store_b64
}

// A fragment: 16x32 bf16, row-major activation tile in LDS (stride ldk).
// Lane L: m = L%16; VGPR0-3 hold K = ks + (L/16)*8 + [0..7],
//         VGPR4-7 hold K = ks + 16 + (L/16)*8 + [0..7].
__device__ inline v16bf load_a(const bf16_t* p, int lane, int ks, int ldk) {
    int m  = lane & 15;
    int kh = (lane >> 4) << 3;
    const bf16_t* q = p + m * ldk + ks + kh;
    v8bf lo = *(const v8bf*)q;
    v8bf hi = *(const v8bf*)(q + 16);
    v16bf r;
#pragma unroll
    for (int i = 0; i < 8; ++i) { r[i] = lo[i]; r[i + 8] = hi[i]; }
    return r;
}

// B fragment: 32x16 bf16, weight stored TRANSPOSED in LDS: wT[n*ldk + k].
// Lane L: n = n0 + L%16; K = ks + (L/16)*8 + [0..7], then ks + 16 + (L/16)*8 + [0..7].
__device__ inline v16bf load_b(const bf16_t* p, int lane, int n0, int ks, int ldk) {
    int n  = n0 + (lane & 15);
    int kb = ks + ((lane >> 4) << 3);
    const bf16_t* q = p + n * ldk + kb;
    v8bf lo = *(const v8bf*)q;
    v8bf hi = *(const v8bf*)(q + 16);
    v16bf r;
#pragma unroll
    for (int i = 0; i < 8; ++i) { r[i] = lo[i]; r[i + 8] = hi[i]; }
    return r;
}

// Cooperative transposed-bf16 weight staging: W is [Krows x 128] row-major f32.
__device__ inline void load_wT128(bf16_t* dst, const float* W, int t, int Krows, int dstStride) {
    int total = Krows * 128;
    for (int idx = t; idx < total; idx += 256) {
        int k = idx >> 7, n = idx & 127;
        dst[n * dstStride + k] = f2bf(W[idx]);
    }
}

__device__ inline v8f wmma_bf16(v16bf a, v16bf b, v8f c) {
    return __builtin_amdgcn_wmma_f32_16x16x32_bf16(false, a, false, b, (short)0, c, false, false);
}

// GEMM: act (16x128 bf16 LDS, stride LDK) x wT (LDS) -> global [.,128] rows r0..r0+15.
__device__ inline void gemm_to_global(const bf16_t* act, const bf16_t* wT, int lane, int wv,
                                      const float* bias, float* gout, int r0) {
    int n0 = wv * 16;
    v8f c = {};
#pragma unroll
    for (int ks = 0; ks < 128; ks += 32)
        c = wmma_bf16(load_a(act, lane, ks, LDK), load_b(wT, lane, n0, ks, LDK), c);
    int n = n0 + (lane & 15), m_off = (lane >> 4) << 3;
#pragma unroll
    for (int e = 0; e < 8; ++e) {
        float v = c[e] + (bias ? bias[n] : 0.f);
        gout[(r0 + e + m_off) * Dq + n] = v;
    }
}

__device__ inline void gemm_to_lds_relu(const bf16_t* act, const bf16_t* wT, int lane, int wv,
                                        const float* bias, bf16_t* out) {
    int n0 = wv * 16;
    v8f c = {};
#pragma unroll
    for (int ks = 0; ks < 128; ks += 32)
        c = wmma_bf16(load_a(act, lane, ks, LDK), load_b(wT, lane, n0, ks, LDK), c);
    int n = n0 + (lane & 15), m_off = (lane >> 4) << 3;
#pragma unroll
    for (int e = 0; e < 8; ++e)
        out[(e + m_off) * LDK + n] = f2bf(fmaxf(c[e] + bias[n], 0.f));
}

// ---------------------------------------------------------------------------
// Kernel 1: per-row stage. delta_self = ReLU(LN(h)@ws1+bs1)@ws2+bs2
//           s1[tau] = h@wi1[tau][:D];  r1[tau] = h@wi1[tau][D:] + bi1[tau]
// ---------------------------------------------------------------------------
__global__ __launch_bounds__(256)
void row_stage(const float* __restrict__ h,
               const float* __restrict__ ls_g, const float* __restrict__ ls_b,
               const float* __restrict__ ws1, const float* __restrict__ bs1,
               const float* __restrict__ ws2, const float* __restrict__ bs2,
               const float* __restrict__ wi1, const float* __restrict__ bi1,
               float* __restrict__ dself, float* __restrict__ s1, float* __restrict__ r1) {
    __shared__ __align__(16) float  hf[16 * 132];
    __shared__ __align__(16) bf16_t aH[16 * LDK];
    __shared__ __align__(16) bf16_t aLN[16 * LDK];
    __shared__ __align__(16) bf16_t mid[16 * LDK];
    __shared__ __align__(16) bf16_t wT[128 * LDK];

    const int t = threadIdx.x, lane = t & 31, wv = t >> 5;
    const int r0 = blockIdx.x * 16;

    // warm L2->WGP path for upcoming weight matrices (global_prefetch_b8)
    __builtin_prefetch(ws1 + t * 64, 0, 1);
    __builtin_prefetch(ws2 + t * 64, 0, 1);

    // stage h rows into LDS (f32)
    {
        int row = t >> 4, c0 = (t & 15) << 3;
#pragma unroll
        for (int u = 0; u < 8; ++u)
            hf[row * 132 + c0 + u] = h[(r0 + row) * Dq + c0 + u];
    }
    __syncthreads();

    // LayerNorm per row -> aLN (bf16); raw h -> aH (bf16). Wave wv: rows 2wv, 2wv+1.
#pragma unroll
    for (int rs = 0; rs < 2; ++rs) {
        int row = wv * 2 + rs;
        float v[4], s = 0.f;
#pragma unroll
        for (int u = 0; u < 4; ++u) { v[u] = hf[row * 132 + lane * 4 + u]; s += v[u]; }
        for (int off = 16; off; off >>= 1) s += __shfl_xor(s, off);
        float mu = s * (1.f / 128.f), vs = 0.f;
#pragma unroll
        for (int u = 0; u < 4; ++u) { float d = v[u] - mu; vs += d * d; }
        for (int off = 16; off; off >>= 1) vs += __shfl_xor(vs, off);
        float rstd = rsqrtf(vs * (1.f / 128.f) + 1e-5f);
        float lnv[4];
#pragma unroll
        for (int u = 0; u < 4; ++u) {
            int col = lane * 4 + u;
            lnv[u] = (v[u] - mu) * rstd * ls_g[col] + ls_b[col];
        }
        store4_bf16(&aLN[row * LDK + lane * 4], lnv);
        store4_bf16(&aH[row * LDK + lane * 4], v);
    }
    __syncthreads();

    // f_self
    load_wT128(wT, ws1, t, 128, LDK); __syncthreads();
    gemm_to_lds_relu(aLN, wT, lane, wv, bs1, mid); __syncthreads();
    load_wT128(wT, ws2, t, 128, LDK); __syncthreads();
    gemm_to_global(mid, wT, lane, wv, bs2, dself, r0); __syncthreads();

    // s1 / r1 per edge type
    for (int tau = 0; tau < Tq; ++tau) {
        const float* w1 = wi1 + tau * (2 * Dq) * Hq;
        __builtin_prefetch(w1 + Dq * Hq + t * 64, 0, 1);
        load_wT128(wT, w1, t, 128, LDK); __syncthreads();
        gemm_to_global(aH, wT, lane, wv, nullptr, s1 + tau * ROWS * Hq, r0); __syncthreads();
        load_wT128(wT, w1 + Dq * Hq, t, 128, LDK); __syncthreads();
        gemm_to_global(aH, wT, lane, wv, bi1 + tau * Hq, r1 + tau * ROWS * Hq, r0); __syncthreads();
    }
}

// ---------------------------------------------------------------------------
// Kernel 2: fused pair stage. For block (jtile, tau, b):
//   y[b,j,:] = sum_i w[b,i,j] * ReLU( ReLU(s1[b,i]+r1[b,j]) @ wi2[tau] + bi2[tau] )
//   sumw[b,j] = sum_i w[b,i,j]
// ---------------------------------------------------------------------------
__global__ __launch_bounds__(256)
void pair_stage(const float* __restrict__ s1, const float* __restrict__ r1,
                const float* __restrict__ edge_probs, const float* __restrict__ edge_types,
                const float* __restrict__ wi2, const float* __restrict__ bi2,
                float* __restrict__ ybuf, float* __restrict__ sumw) {
    __shared__ __align__(16) bf16_t wT[128 * LDK];     // wi2[tau] transposed
    __shared__ __align__(16) float  r1s[16 * 132];     // receiver tile (incl. bi1)
    __shared__ __align__(16) float  s1s[16 * 132];     // sender tile
    __shared__ __align__(16) bf16_t x1s[16 * LDK];     // ReLU(s1+r1) tile, bf16
    __shared__ __align__(16) float  wgt[64 * 16];      // edge weights [i][jl]
    __shared__ __align__(16) float  bi2s[128];

    const int jt = blockIdx.x, tau = blockIdx.y, b = blockIdx.z;
    const int t = threadIdx.x, lane = t & 31, wv = t >> 5;
    const int j0 = jt * 16;
    const int rowbase = tau * ROWS + b * Kq;

    load_wT128(wT, wi2 + tau * Hq * Hq, t, 128, LDK);
    for (int idx = t; idx < 16 * 128; idx += 256) {
        int rr = idx >> 7, cc = idx & 127;
        r1s[rr * 132 + cc] = r1[(rowbase + j0 + rr) * Hq + cc];
    }
    for (int idx = t; idx < 64 * 16; idx += 256) {
        int i = idx >> 4, jl = idx & 15, j = j0 + jl;
        float p  = edge_probs[(b * Kq + i) * Kq + j];
        float et = edge_types[(((b * Kq + i) * Kq + j) * Tq) + tau];
        wgt[i * 16 + jl] = p * et;
    }
    if (t < 128) bi2s[t] = bi2[tau * Hq + t];
    __syncthreads();

    float yacc[16];
#pragma unroll
    for (int j = 0; j < 16; ++j) yacc[j] = 0.f;

    const int n0 = wv * 16;
    const int m_off = (lane >> 4) << 3;
    const int xrow = t >> 4, xc0 = (t & 15) << 3;

    for (int it = 0; it < 4; ++it) {                  // sender tiles of 16
        // prefetch the next sender tile while this one is consumed
        if (it < 3) __builtin_prefetch(s1 + (rowbase + (it + 1) * 16) * Hq + t * 8, 0, 1);
        for (int idx = t; idx < 16 * 128; idx += 256) {
            int rr = idx >> 7, cc = idx & 127;
            s1s[rr * 132 + cc] = s1[(rowbase + it * 16 + rr) * Hq + cc];
        }
        __syncthreads();
        for (int jl = 0; jl < 16; ++jl) {             // receivers in this tile
            {   // build x1 = ReLU(s1[i] + r1[jl]) -> bf16, cooperative, one b128 store
                float tmp[8];
#pragma unroll
                for (int u = 0; u < 8; ++u)
                    tmp[u] = fmaxf(s1s[xrow * 132 + xc0 + u] + r1s[jl * 132 + xc0 + u], 0.f);
                store8_bf16(&x1s[xrow * LDK + xc0], tmp);
            }
            __syncthreads();
            v8f c = {};
#pragma unroll
            for (int ks = 0; ks < 128; ks += 32)
                c = wmma_bf16(load_a(x1s, lane, ks, LDK), load_b(wT, lane, n0, ks, LDK), c);
            // fused epilogue: ReLU + weighted reduction over the 16 sender rows
            int n = n0 + (lane & 15);
            float p = 0.f;
#pragma unroll
            for (int e = 0; e < 8; ++e) {
                float x2 = fmaxf(c[e] + bi2s[n], 0.f);
                p += wgt[(it * 16 + e + m_off) * 16 + jl] * x2;
            }
            p += __shfl_xor(p, 16);                   // combine lane halves (i rows 0-7 / 8-15)
            yacc[jl] += p;
            __syncthreads();
        }
    }

    // writeback y (lane 0..15 hold n = n0+lane)
    for (int jl = 0; jl < 16; ++jl)
        if (lane < 16)
            ybuf[(rowbase + j0 + jl) * Hq + n0 + lane] = yacc[jl];

    // sumw: wave wv handles jl = wv and wv+8
#pragma unroll
    for (int js = 0; js < 2; ++js) {
        int jl = wv + js * 8;
        float s = wgt[lane * 16 + jl] + wgt[(lane + 32) * 16 + jl];
        for (int off = 16; off; off >>= 1) s += __shfl_xor(s, off);
        if (lane == 0) sumw[rowbase + j0 + jl] = s;
    }
}

// ---------------------------------------------------------------------------
// Kernel 3: delta_inter[r] = sum_tau ( y[tau][r] @ wi3[tau] + sumw[tau][r]*bi3[tau] )
// ---------------------------------------------------------------------------
__global__ __launch_bounds__(256)
void inter_stage(const float* __restrict__ ybuf, const float* __restrict__ sumw,
                 const float* __restrict__ wi3, const float* __restrict__ bi3,
                 float* __restrict__ dinter) {
    __shared__ __align__(16) bf16_t wT[128 * LDK];
    __shared__ __align__(16) bf16_t act[16 * LDK];

    const int t = threadIdx.x, lane = t & 31, wv = t >> 5;
    const int r0 = blockIdx.x * 16;
    const int n0 = wv * 16;
    const int arow = t >> 4, ac0 = (t & 15) << 3;

    v8f c = {};
    for (int tau = 0; tau < Tq; ++tau) {
        {   // stage y rows as bf16, vectorized
            float tmp[8];
            const float* src = &ybuf[(tau * ROWS + r0 + arow) * Hq + ac0];
#pragma unroll
            for (int u = 0; u < 8; ++u) tmp[u] = src[u];
            store8_bf16(&act[arow * LDK + ac0], tmp);
        }
        load_wT128(wT, wi3 + tau * Hq * Dq, t, 128, LDK);
        __syncthreads();
#pragma unroll
        for (int ks = 0; ks < 128; ks += 32)
            c = wmma_bf16(load_a(act, lane, ks, LDK), load_b(wT, lane, n0, ks, LDK), c);
        __syncthreads();
    }
    int n = n0 + (lane & 15), m_off = (lane >> 4) << 3;
#pragma unroll
    for (int e = 0; e < 8; ++e) {
        int row = r0 + e + m_off;
        float bias = 0.f;
#pragma unroll
        for (int tau = 0; tau < Tq; ++tau)
            bias += sumw[tau * ROWS + row] * bi3[tau * Dq + n];
        dinter[row * Dq + n] = c[e] + bias;
    }
}

// ---------------------------------------------------------------------------
// Kernel 4: update.  combined = [h+dself, dinter] (2D=256); LN(256);
//           h_out = h + ReLU(LN@wu1+bu1)@wu2 + bu2
// ---------------------------------------------------------------------------
__global__ __launch_bounds__(256)
void update_stage(const float* __restrict__ h, const float* __restrict__ dself,
                  const float* __restrict__ dinter,
                  const float* __restrict__ lu_g, const float* __restrict__ lu_b,
                  const float* __restrict__ wu1, const float* __restrict__ bu1,
                  const float* __restrict__ wu2, const float* __restrict__ bu2,
                  float* __restrict__ hout) {
    __shared__ __align__(16) float  cf[16 * 260];      // combined f32
    __shared__ __align__(16) bf16_t cln[16 * 264];     // LN'd combined bf16 (stride 264)
    __shared__ __align__(16) bf16_t wT[128 * LDK];     // one 128-K half of a weight
    __shared__ __align__(16) bf16_t mid[16 * LDK];

    const int t = threadIdx.x, lane = t & 31, wv = t >> 5;
    const int r0 = blockIdx.x * 16;
    const int n0 = wv * 16;

    __builtin_prefetch(wu1 + t * 128, 0, 1);

    for (int idx = t; idx < 16 * 256; idx += 256) {
        int rr = idx >> 8, cc = idx & 255;
        float v = (cc < 128) ? h[(r0 + rr) * Dq + cc] + dself[(r0 + rr) * Dq + cc]
                             : dinter[(r0 + rr) * Dq + (cc - 128)];
        cf[rr * 260 + cc] = v;
    }
    __syncthreads();

    // LayerNorm over 256: wave wv handles rows 2wv, 2wv+1; lane covers 8 cols
#pragma unroll
    for (int rs = 0; rs < 2; ++rs) {
        int row = wv * 2 + rs;
        float v[8], s = 0.f;
#pragma unroll
        for (int u = 0; u < 8; ++u) { v[u] = cf[row * 260 + lane * 8 + u]; s += v[u]; }
        for (int off = 16; off; off >>= 1) s += __shfl_xor(s, off);
        float mu = s * (1.f / 256.f), vs = 0.f;
#pragma unroll
        for (int u = 0; u < 8; ++u) { float d = v[u] - mu; vs += d * d; }
        for (int off = 16; off; off >>= 1) vs += __shfl_xor(vs, off);
        float rstd = rsqrtf(vs * (1.f / 256.f) + 1e-5f);
        float lnv[8];
#pragma unroll
        for (int u = 0; u < 8; ++u) {
            int col = lane * 8 + u;
            lnv[u] = (v[u] - mu) * rstd * lu_g[col] + lu_b[col];
        }
        store8_bf16(&cln[row * 264 + lane * 8], lnv);
    }
    __syncthreads();

    // GEMM1: K=256 in two 128-K halves accumulating into c
    v8f c = {};
    load_wT128(wT, wu1, t, 128, LDK); __syncthreads();
#pragma unroll
    for (int ks = 0; ks < 128; ks += 32)
        c = wmma_bf16(load_a(cln, lane, ks, 264), load_b(wT, lane, n0, ks, LDK), c);
    __syncthreads();
    load_wT128(wT, wu1 + 128 * Hq, t, 128, LDK); __syncthreads();
#pragma unroll
    for (int ks = 0; ks < 128; ks += 32)
        c = wmma_bf16(load_a(cln, lane, 128 + ks, 264), load_b(wT, lane, n0, ks, LDK), c);

    int n = n0 + (lane & 15), m_off = (lane >> 4) << 3;
#pragma unroll
    for (int e = 0; e < 8; ++e)
        mid[(e + m_off) * LDK + n] = f2bf(fmaxf(c[e] + bu1[n], 0.f));
    __syncthreads();

    // GEMM2 + residual
    load_wT128(wT, wu2, t, 128, LDK); __syncthreads();
    v8f c2 = {};
#pragma unroll
    for (int ks = 0; ks < 128; ks += 32)
        c2 = wmma_bf16(load_a(mid, lane, ks, LDK), load_b(wT, lane, n0, ks, LDK), c2);
#pragma unroll
    for (int e = 0; e < 8; ++e) {
        int row = r0 + e + m_off;
        hout[row * Dq + n] = h[row * Dq + n] + c2[e] + bu2[n];
    }
}

// ---------------------------------------------------------------------------
extern "C" void kernel_launch(void* const* d_in, const int* in_sizes, int n_in,
                              void* d_out, int out_size, void* d_ws, size_t ws_size,
                              hipStream_t stream) {
    (void)in_sizes; (void)n_in; (void)out_size; (void)ws_size;
    const float* slots      = (const float*)d_in[0];
    const float* edge_probs = (const float*)d_in[1];
    const float* edge_types = (const float*)d_in[2];
    const float* ls_g = (const float*)d_in[3];
    const float* ls_b = (const float*)d_in[4];
    const float* ws1  = (const float*)d_in[5];
    const float* bs1  = (const float*)d_in[6];
    const float* ws2  = (const float*)d_in[7];
    const float* bs2  = (const float*)d_in[8];
    const float* wi1  = (const float*)d_in[9];
    const float* bi1  = (const float*)d_in[10];
    const float* wi2  = (const float*)d_in[11];
    const float* bi2  = (const float*)d_in[12];
    const float* wi3  = (const float*)d_in[13];
    const float* bi3  = (const float*)d_in[14];
    const float* lu_g = (const float*)d_in[15];
    const float* lu_b = (const float*)d_in[16];
    const float* wu1  = (const float*)d_in[17];
    const float* bu1  = (const float*)d_in[18];
    const float* wu2  = (const float*)d_in[19];
    const float* bu2  = (const float*)d_in[20];

    char* ws = (char*)d_ws;
    size_t off = 0;
    auto salloc = [&](size_t bytes) { void* p = ws + off; off += (bytes + 255) & ~(size_t)255; return p; };
    float* dself  = (float*)salloc((size_t)ROWS * Dq * 4);
    float* s1     = (float*)salloc((size_t)Tq * ROWS * Hq * 4);
    float* r1     = (float*)salloc((size_t)Tq * ROWS * Hq * 4);
    float* ybuf   = (float*)salloc((size_t)Tq * ROWS * Hq * 4);
    float* sumw   = (float*)salloc((size_t)Tq * ROWS * 4);
    float* dinter = (float*)salloc((size_t)ROWS * Dq * 4);
    float* hA     = (float*)salloc((size_t)ROWS * Dq * 4);

    const float* hin = slots;
    for (int mp = 0; mp < 2; ++mp) {
        float* hout = (mp == 1) ? (float*)d_out : hA;
        row_stage<<<ROWS / 16, 256, 0, stream>>>(hin, ls_g, ls_b, ws1, bs1, ws2, bs2,
                                                 wi1, bi1, dself, s1, r1);
        pair_stage<<<dim3(Kq / 16, Tq, Bq), 256, 0, stream>>>(s1, r1, edge_probs, edge_types,
                                                              wi2, bi2, ybuf, sumw);
        inter_stage<<<ROWS / 16, 256, 0, stream>>>(ybuf, sumw, wi3, bi3, dinter);
        update_stage<<<ROWS / 16, 256, 0, stream>>>(hin, dself, dinter, lu_g, lu_b,
                                                    wu1, bu1, wu2, bu2, hout);
        hin = hout;
    }
}
